// TemporalRelation_15951508537871
// MI455X (gfx1250) — matrix operand order repaired and checked by reading, not compiled, not run on hardware
//
#include <hip/hip_runtime.h>

typedef __attribute__((ext_vector_type(16))) _Float16 v16h;
typedef __attribute__((ext_vector_type(8)))  _Float16 v8h;
typedef __attribute__((ext_vector_type(8)))  float    v8f;

// LDS layout (in halves)
#define ACT1_OFF 0                  // 130 rows x 512 ch  = 66560 halves (conv1 out, row = pos+1)
#define ACT2_OFF 66560              // 130 rows x 256 ch  = 33280 halves (conv2 out)
#define BCOL_OFF 99840              // 128 pos x 32 K     =  4096 halves (im2col of 4-ch rel, K padded 12->32)
#define SMEM_HALVES 103936          // 207,872 bytes < 320KB WGP LDS

__device__ static inline v16h cat8(v8h lo, v8h hi) {
    return __builtin_shufflevector(lo, hi, 0, 1, 2, 3, 4, 5, 6, 7, 8, 9, 10, 11, 12, 13, 14, 15);
}

// ---- weight prep: transpose to [co][dk*Cin+ci] and convert to f16 ----
__global__ void prep_w1_kernel(const float* __restrict__ W1, _Float16* __restrict__ A1) {
    int idx = blockIdx.x * blockDim.x + threadIdx.x;       // 512*32
    if (idx >= 512 * 32) return;
    int co = idx >> 5, k = idx & 31;
    float v = 0.f;
    if (k < 12) { int dk = k >> 2, ci = k & 3; v = W1[(co * 4 + ci) * 3 + dk]; }
    A1[idx] = (_Float16)v;
}

__global__ void prep_w2_kernel(const float* __restrict__ W2, _Float16* __restrict__ A2) {
    int idx = blockIdx.x * blockDim.x + threadIdx.x;       // 256*1536
    if (idx >= 256 * 1536) return;
    int co = idx / 1536, k = idx - co * 1536;
    int dk = k >> 9, ci = k & 511;
    A2[idx] = (_Float16)W2[(co * 512 + ci) * 3 + dk];
}

// ---- fused 3-layer temporal-relation conv, one sample per workgroup ----
__global__ __launch_bounds__(256, 1)
void fused_trn_kernel(const float* __restrict__ history, const float* __restrict__ current,
                      const _Float16* __restrict__ A1, const float* __restrict__ b1,
                      const _Float16* __restrict__ A2, const float* __restrict__ b2,
                      const float* __restrict__ W3, const float* __restrict__ b3,
                      float* __restrict__ out)
{
    extern __shared__ _Float16 smem[];
    _Float16* act1 = smem + ACT1_OFF;
    _Float16* act2 = smem + ACT2_OFF;
    _Float16* bcol = smem + BCOL_OFF;

    const int tid   = threadIdx.x;
    const int lane  = tid & 31;
    const int wave  = tid >> 5;
    const int l15   = lane & 15;
    const int lhalf = lane >> 4;            // 0 or 1

    const int s = blockIdx.x;               // sample = m*32 + b (matches reshape(M*B,...))
    const int m = s >> 5;
    const int b = s & 31;
    const float* hrow = history + (size_t)(b * 64 + m) * 128;
    const float* crow = current + (size_t)b * 128;

    // ---------- stage 0: zero padding rows; build im2col B for conv1 ----------
    for (int i = tid; i < 512; i += 256) { act1[i] = (_Float16)0.f; act1[129 * 512 + i] = (_Float16)0.f; }
    act2[tid] = (_Float16)0.f;
    act2[129 * 256 + tid] = (_Float16)0.f;
    {
        int n  = tid >> 1;                  // position 0..127
        int k0 = (tid & 1) * 16;
        #pragma unroll
        for (int kk = 0; kk < 16; ++kk) {
            int k = k0 + kk;
            float v = 0.f;
            if (k < 12) {
                int dk = k >> 2, ci = k & 3;
                int p = n + dk - 1;
                if (p >= 0 && p < 128) {
                    float h = hrow[p], c = crow[p];
                    v = (ci == 0) ? h : (ci == 1) ? c : (ci == 2) ? (h - c) : (c - h);
                }
            }
            bcol[n * 32 + k] = (_Float16)v;
        }
    }
    __syncthreads();

    // ---------- stage 1: conv1  (512x128 out, 32 Mt x 8 Nt tiles, one K=32 step) ----------
    // t = wave + 8*i  =>  nt = wave is loop-invariant (B fragment hoists out of the loop)
    #pragma unroll 1
    for (int t = wave; t < 256; t += 8) {
        int mt = t >> 3, nt = t & 7;
        int mrow = mt * 16 + l15;
        const v8h* ap = (const v8h*)(A1 + mrow * 32 + lhalf * 8);
        v8h alo = ap[0], ahi = ap[2];                       // K = khalf..+7 and 16+khalf..+7
        int nn = nt * 16 + l15;
        const v8h* bp = (const v8h*)(bcol + nn * 32 + lhalf * 8);
        v8h blo = bp[0], bhi = bp[2];                       // K = khalf..+7 and 16+khalf..+7
        v16h a  = cat8(alo, ahi);
        v16h bb = cat8(blo, bhi);
        v8f c = {};
        c = __builtin_amdgcn_wmma_f32_16x16x32_f16(false, a, false, bb, (short)0, c, false, false);
        int colbase = mt * 16 + lhalf * 8;                  // channel base for this lane's 8 rows
        v8h o;
        #pragma unroll
        for (int r = 0; r < 8; ++r) o[r] = (_Float16)(c[r] + b1[colbase + r]);
        *(v8h*)(act1 + (nn + 1) * 512 + colbase) = o;
    }
    __syncthreads();

    // ---------- stage 2: conv2  (256x128 out, K=1536; per wave: 2 Mt x 4 Nt blocks) ----------
    // 16 blocks total (8 co-blocks of 32 x 2 pos-blocks of 64), 2 per wave.
    // Per 32-K step: 2 global A-frags + 4 LDS B-frags feed 8 WMMAs (0.75 loads/WMMA).
    #pragma unroll 1
    for (int blk = wave; blk < 16; blk += 8) {
        int mb = blk >> 1;                  // 0..7 -> co base mb*32
        int nb = blk & 1;                   // 0..1 -> pos base nb*64
        v8f acc[2][4];
        #pragma unroll
        for (int i = 0; i < 2; ++i)
            #pragma unroll
            for (int j = 0; j < 4; ++j) acc[i][j] = (v8f){};
        const int arow0 = (mb * 32 + l15) * 1536;
        const int arow1 = (mb * 32 + 16 + l15) * 1536;
        const int nbase = nb * 64 + l15;    // + j*16 per N-tile
        #pragma unroll 2
        for (int ks = 0; ks < 48; ++ks) {
            int kbase = ks * 32;
            int dk = kbase >> 9;                            // constant within 32-K block
            int ci = (kbase & 511) + lhalf * 8;
            int khalf = lhalf * 8;
            const v8h* ap0 = (const v8h*)(A2 + arow0 + kbase + khalf);
            const v8h* ap1 = (const v8h*)(A2 + arow1 + kbase + khalf);
            v16h a0 = cat8(ap0[0], ap0[2]);
            v16h a1 = cat8(ap1[0], ap1[2]);
            v16h bbv[4];
            #pragma unroll
            for (int j = 0; j < 4; ++j) {
                const v8h* bp = (const v8h*)(act1 + (nbase + j * 16 + dk) * 512 + ci);
                bbv[j] = cat8(bp[0], bp[2]);
            }
            #pragma unroll
            for (int j = 0; j < 4; ++j) {
                acc[0][j] = __builtin_amdgcn_wmma_f32_16x16x32_f16(false, a0, false, bbv[j], (short)0, acc[0][j], false, false);
                acc[1][j] = __builtin_amdgcn_wmma_f32_16x16x32_f16(false, a1, false, bbv[j], (short)0, acc[1][j], false, false);
            }
        }
        // bias + f16 convert + store to act2 (row = pos+1, col = channel)
        #pragma unroll
        for (int i = 0; i < 2; ++i) {
            int col = (mb * 2 + i) * 16 + lhalf * 8;
            #pragma unroll
            for (int j = 0; j < 4; ++j) {
                int row = (nb * 4 + j) * 16 + l15 + 1;
                v8h o;
                #pragma unroll
                for (int r = 0; r < 8; ++r) o[r] = (_Float16)(acc[i][j][r] + b2[col + r]);
                *(v8h*)(act2 + row * 256 + col) = o;
            }
        }
    }
    __syncthreads();

    // ---------- stage 3: conv3 (256 -> 1), cheap VALU reduction, W3 lane-invariant ----------
    if (tid < 128) {
        int n = tid;
        float acc = b3[0];
        #pragma unroll
        for (int dk = 0; dk < 3; ++dk) {
            const _Float16* arow = act2 + (n + dk) * 256;
            #pragma unroll 4
            for (int c8 = 0; c8 < 256; c8 += 8) {
                v8h av = *(const v8h*)(arow + c8);
                #pragma unroll
                for (int j = 0; j < 8; ++j)
                    acc += (float)av[j] * W3[(c8 + j) * 3 + dk];
            }
        }
        out[(size_t)s * 128 + n] = acc;
    }
}

extern "C" void kernel_launch(void* const* d_in, const int* in_sizes, int n_in,
                              void* d_out, int out_size, void* d_ws, size_t ws_size,
                              hipStream_t stream) {
    const float* history = (const float*)d_in[0];
    const float* current = (const float*)d_in[1];
    const float* W1 = (const float*)d_in[2];
    const float* b1 = (const float*)d_in[3];
    const float* W2 = (const float*)d_in[4];
    const float* b2 = (const float*)d_in[5];
    const float* W3 = (const float*)d_in[6];
    const float* b3 = (const float*)d_in[7];
    float* out = (float*)d_out;

    _Float16* A1 = (_Float16*)d_ws;                           // 512*32 halves   = 32 KB
    _Float16* A2 = (_Float16*)((char*)d_ws + 64 * 1024);      // 256*1536 halves = 768 KB

    prep_w1_kernel<<<(512 * 32 + 255) / 256, 256, 0, stream>>>(W1, A1);
    prep_w2_kernel<<<(256 * 1536 + 255) / 256, 256, 0, stream>>>(W2, A2);

    const size_t smem_bytes = (size_t)SMEM_HALVES * sizeof(_Float16);   // ~203 KB dynamic LDS
    fused_trn_kernel<<<2048, 256, smem_bytes, stream>>>(history, current, A1, b1, A2, b2, W3, b3, out);
}
